// FeatureNet_72258529788150
// MI455X (gfx1250) — matrix operand order repaired and checked by reading, not compiled, not run on hardware
//
#include <hip/hip_runtime.h>

typedef __attribute__((ext_vector_type(16))) __bf16 v16bf;
typedef __attribute__((ext_vector_type(8)))  __bf16 v8bf;
typedef __attribute__((ext_vector_type(8)))  float  v8f;

#define T_STEPS 1000
#define BATCH   256
#define DIN     64
#define HID     128
#define G4      512            // 4*H gate width
#define BW      16             // batch rows per workgroup
#define NWG     (BATCH / BW)
#define WIH_S   72             // bf16 row stride for x tile (36 dwords: conflict-free)
#define WHH_S   136            // bf16 row stride for h tile (68 dwords: conflict-free)
#define GS      516            // f32 row stride for gate buffers

// A operand (16x32, MxK) from LDS: lane group hi=lane>>4; per lane 16 contiguous K
// at K = koff + 16*hi (matches B contiguous order)
__device__ __forceinline__ v16bf ldA(const __bf16* p) {
  union { v16bf v; v8bf h[2]; } u;
  u.h[0] = *(const v8bf*)(p);
  u.h[1] = *(const v8bf*)(p + 8);
  return u.v;
}

// B operand slot from GLOBAL f32: 16 consecutive K values of one weight row,
// converted to bf16. (B 32x16: column n = lane&15, contiguous 16 K at 16*hi.)
__device__ __forceinline__ v16bf ldBg(const float* p) {
  const float4* q = (const float4*)p;
  float4 a = q[0], b = q[1], c = q[2], d = q[3];
  v16bf r;
  r[0]  = (__bf16)a.x; r[1]  = (__bf16)a.y; r[2]  = (__bf16)a.z; r[3]  = (__bf16)a.w;
  r[4]  = (__bf16)b.x; r[5]  = (__bf16)b.y; r[6]  = (__bf16)b.z; r[7]  = (__bf16)b.w;
  r[8]  = (__bf16)c.x; r[9]  = (__bf16)c.y; r[10] = (__bf16)c.z; r[11] = (__bf16)c.w;
  r[12] = (__bf16)d.x; r[13] = (__bf16)d.y; r[14] = (__bf16)d.z; r[15] = (__bf16)d.w;
  return r;
}

__device__ __forceinline__ v8f wmma_bf16(v16bf a, v16bf b, v8f c) {
  return __builtin_amdgcn_wmma_f32_16x16x32_bf16(false, a, false, b, (short)0, c,
                                                 false, false);
}

__device__ __forceinline__ float wsum32(float v) {
#pragma unroll
  for (int m = 16; m; m >>= 1) v += __shfl_xor(v, m, 32);
  return v;
}

__device__ __forceinline__ float sigm(float x) {
  return __builtin_amdgcn_rcpf(1.0f + __expf(-x));
}
__device__ __forceinline__ float tanh_f(float x) {
#if __has_builtin(__builtin_amdgcn_tanhf)
  return __builtin_amdgcn_tanhf(x);
#else
  return 2.0f * __builtin_amdgcn_rcpf(1.0f + __expf(-2.0f * x)) - 1.0f;
#endif
}

__global__ __launch_bounds__(256, 1)
void lnlstm_persistent(const float* __restrict__ x, const float* __restrict__ h0,
                       const float* __restrict__ c0, const float* __restrict__ Wih,
                       const float* __restrict__ Whh, const float* __restrict__ gi,
                       const float* __restrict__ bi, const float* __restrict__ ghp,
                       const float* __restrict__ bhp, const float* __restrict__ gc,
                       const float* __restrict__ bc, float* __restrict__ ys,
                       float* __restrict__ hT, float* __restrict__ cT) {
  // ---- LDS (~98 KB): activations + state only; weights live in VGPRs ----
  __shared__ __bf16 s_x[BW * WIH_S];     //  2,304 B  x_t tile bf16
  __shared__ __bf16 s_h[BW * WHH_S];     //  4,352 B  h tile bf16
  __shared__ float  s_ga[BW * GS];       // 33,024 B  x @ Wih^T (f32)
  __shared__ float  s_gb[BW * GS];       // 33,024 B  h @ Whh^T (f32)
  __shared__ float  s_c[BW * HID];       //  8,192 B  cell state f32
  __shared__ float  s_o[BW * HID];       //  8,192 B  sigmoid(o)
  __shared__ float  s_pgi[G4], s_pbi[G4], s_pgh[G4], s_pbh[G4];
  __shared__ float  s_pgc[HID], s_pbc[HID];
  __shared__ float  s_m1[BW], s_r1[BW], s_m2[BW], s_r2[BW], s_mc[BW], s_rc[BW];

  const int tid  = threadIdx.x;
  const int lane = tid & 31;
  const int wave = tid >> 5;
  const int row  = lane & 15;
  const int hi   = lane >> 4;
  const int b0   = blockIdx.x * BW;

  // ---- persistent weight registers: B operands for this wave's 4 tiles ----
  // wih_r: 4 tiles x 2 K-steps (K=64); whh_r: 4 tiles x 4 K-steps (K=128)
  v16bf wih_r[4][2];
  v16bf whh_r[4][4];
#pragma unroll
  for (int q = 0; q < 4; ++q) {
    const int ncol = (wave * 4 + q) * 16 + row;   // weight row == output column
#pragma unroll
    for (int kk = 0; kk < 2; ++kk)
      wih_r[q][kk] = ldBg(Wih + (size_t)ncol * DIN + kk * 32 + hi * 16);
#pragma unroll
    for (int kk = 0; kk < 4; ++kk)
      whh_r[q][kk] = ldBg(Whh + (size_t)ncol * HID + kk * 32 + hi * 16);
  }

  // ---- one-time: params + h0/c0 slice -> LDS ----
  for (int i = tid; i < G4; i += 256) {
    s_pgi[i] = gi[i]; s_pbi[i] = bi[i]; s_pgh[i] = ghp[i]; s_pbh[i] = bhp[i];
  }
  if (tid < HID) { s_pgc[tid] = gc[tid]; s_pbc[tid] = bc[tid]; }
  for (int i = tid; i < BW * HID / 4; i += 256) {
    float4 h4 = ((const float4*)(h0 + (size_t)b0 * HID))[i];
    int lin = i << 2, r = lin >> 7, cc = lin & 127;
    __bf16* p = &s_h[r * WHH_S + cc];
    p[0] = (__bf16)h4.x; p[1] = (__bf16)h4.y; p[2] = (__bf16)h4.z; p[3] = (__bf16)h4.w;
    float4 c4 = ((const float4*)(c0 + (size_t)b0 * HID))[i];
    *(float4*)&s_c[lin] = c4;
  }

  // ---- register double-buffer for the x tile (tile t is contiguous 4 KB) ----
  float4 xreg = ((const float4*)(x + (size_t)b0 * DIN))[tid];   // t = 0
  __syncthreads();

  for (int t = 0; t < T_STEPS; ++t) {
    // ---- commit x_t (held in regs since last step) -> bf16 LDS ----
    {
      int lin = tid << 2, r = lin >> 6, cc = lin & 63;
      __bf16* p = &s_x[r * WIH_S + cc];
      p[0] = (__bf16)xreg.x; p[1] = (__bf16)xreg.y;
      p[2] = (__bf16)xreg.z; p[3] = (__bf16)xreg.w;
    }
    __syncthreads();

    // issue next tile's load now; the wait lands at the top of the next step,
    // hidden behind this entire step's compute
    if (t + 1 < T_STEPS)
      xreg = ((const float4*)(x + ((size_t)(t + 1) * BATCH + b0) * DIN))[tid];

    // ---- GEMMs: A from LDS (issued up front), B from registers, no stalls ----
    {
      v16bf ax0 = ldA(&s_x[row * WIH_S +  0 + hi * 16]);
      v16bf ax1 = ldA(&s_x[row * WIH_S + 32 + hi * 16]);
      v16bf ah0 = ldA(&s_h[row * WHH_S +  0 + hi * 16]);
      v16bf ah1 = ldA(&s_h[row * WHH_S + 32 + hi * 16]);
      v16bf ah2 = ldA(&s_h[row * WHH_S + 64 + hi * 16]);
      v16bf ah3 = ldA(&s_h[row * WHH_S + 96 + hi * 16]);
#pragma unroll
      for (int q = 0; q < 4; ++q) {
        const int n = (wave * 4 + q) * 16 + row;   // C: N = lane&15, M = r + 8*hi
        v8f acc = {0.f, 0.f, 0.f, 0.f, 0.f, 0.f, 0.f, 0.f};
        acc = wmma_bf16(ax0, wih_r[q][0], acc);
        acc = wmma_bf16(ax1, wih_r[q][1], acc);
        v8f acb = {0.f, 0.f, 0.f, 0.f, 0.f, 0.f, 0.f, 0.f};
        acb = wmma_bf16(ah0, whh_r[q][0], acb);
        acb = wmma_bf16(ah1, whh_r[q][1], acb);
        acb = wmma_bf16(ah2, whh_r[q][2], acb);
        acb = wmma_bf16(ah3, whh_r[q][3], acb);
#pragma unroll
        for (int rr = 0; rr < 8; ++rr) {
          s_ga[(hi * 8 + rr) * GS + n] = acc[rr];
          s_gb[(hi * 8 + rr) * GS + n] = acb[rr];
        }
      }
    }
    __syncthreads();

    // ---- LN stats over 512 cols: 32 row-jobs (2 matrices x 16 rows), 4/wave ----
#pragma unroll
    for (int q = 0; q < 4; ++q) {
      const int job = wave * 4 + q;
      const float* src = (job & 16) ? s_gb : s_ga;
      const int r = job & 15;
      const float* p = src + r * GS + lane * 16;
      float s = 0.f, sq = 0.f;
#pragma unroll
      for (int k = 0; k < 16; ++k) { float v = p[k]; s += v; sq = fmaf(v, v, sq); }
      s = wsum32(s); sq = wsum32(sq);
      if (lane == 0) {
        float m  = s * (1.0f / 512.0f);
        float vr = sq * (1.0f / 512.0f) - m * m;
        float rs = rsqrtf(vr + 1e-5f);
        if (job & 16) { s_m2[r] = m; s_r2[r] = rs; }
        else          { s_m1[r] = m; s_r1[r] = rs; }
      }
    }
    __syncthreads();

    // ---- gates (PyTorch order i,f,g,o) + cell update ----
    for (int it = tid; it < BW * HID; it += 256) {
      const int b = it >> 7, j = it & 127;
      const float m1 = s_m1[b], r1 = s_r1[b], m2 = s_m2[b], r2 = s_r2[b];
      const float* pa = &s_ga[b * GS];
      const float* pb = &s_gb[b * GS];
      float zi = (pa[j      ] - m1) * r1 * s_pgi[j      ] + s_pbi[j      ] +
                 (pb[j      ] - m2) * r2 * s_pgh[j      ] + s_pbh[j      ];
      float zf = (pa[j + 128] - m1) * r1 * s_pgi[j + 128] + s_pbi[j + 128] +
                 (pb[j + 128] - m2) * r2 * s_pgh[j + 128] + s_pbh[j + 128];
      float zg = (pa[j + 256] - m1) * r1 * s_pgi[j + 256] + s_pbi[j + 256] +
                 (pb[j + 256] - m2) * r2 * s_pgh[j + 256] + s_pbh[j + 256];
      float zo = (pa[j + 384] - m1) * r1 * s_pgi[j + 384] + s_pbi[j + 384] +
                 (pb[j + 384] - m2) * r2 * s_pgh[j + 384] + s_pbh[j + 384];
      float cn = sigm(zf) * s_c[it] + sigm(zi) * tanh_f(zg);
      s_c[it] = cn;
      s_o[it] = sigm(zo);
    }
    __syncthreads();

    // ---- LN stats over c rows (128 wide): 2 rows per wave ----
#pragma unroll
    for (int q = 0; q < 2; ++q) {
      const int r = wave * 2 + q;
      const float* p = &s_c[r * HID + lane * 4];
      float s = 0.f, sq = 0.f;
#pragma unroll
      for (int k = 0; k < 4; ++k) { float v = p[k]; s += v; sq = fmaf(v, v, sq); }
      s = wsum32(s); sq = wsum32(sq);
      if (lane == 0) {
        float m  = s * (1.0f / 128.0f);
        float vr = sq * (1.0f / 128.0f) - m * m;
        s_mc[r] = m;
        s_rc[r] = rsqrtf(vr + 1e-5f);
      }
    }
    __syncthreads();

    // ---- h update: h = sigmoid(o) * tanh(LN(c)); emit ys[t] ----
    for (int it = tid; it < BW * HID; it += 256) {
      const int b = it >> 7, j = it & 127;
      const float cn = s_c[it];
      const float hn =
          s_o[it] * tanh_f((cn - s_mc[b]) * s_rc[b] * s_pgc[j] + s_pbc[j]);
      s_h[b * WHH_S + j] = (__bf16)hn;
      ys[((size_t)t * BATCH + b0 + b) * HID + j] = hn;
      if (t == T_STEPS - 1) hT[(size_t)(b0 + b) * HID + j] = hn;
    }
    __syncthreads();
  }

  // ---- final cell state ----
  for (int it = tid; it < BW * HID; it += 256) {
    const int b = it >> 7, j = it & 127;
    cT[(size_t)(b0 + b) * HID + j] = s_c[it];
  }
}

extern "C" void kernel_launch(void* const* d_in, const int* in_sizes, int n_in,
                              void* d_out, int out_size, void* d_ws, size_t ws_size,
                              hipStream_t stream) {
  const float* x   = (const float*)d_in[0];
  const float* h0  = (const float*)d_in[1];
  const float* c0  = (const float*)d_in[2];
  const float* Wih = (const float*)d_in[3];
  const float* Whh = (const float*)d_in[4];
  const float* gi  = (const float*)d_in[5];
  const float* bi  = (const float*)d_in[6];
  const float* gh  = (const float*)d_in[7];
  const float* bh  = (const float*)d_in[8];
  const float* gc  = (const float*)d_in[9];
  const float* bc  = (const float*)d_in[10];
  float* ys = (float*)d_out;                               // [T, B, H]
  float* hT = ys + (size_t)T_STEPS * BATCH * HID;          // [B, H]
  float* cT = hT + (size_t)BATCH * HID;                    // [B, H]
  lnlstm_persistent<<<NWG, 256, 0, stream>>>(x, h0, c0, Wih, Whh, gi, bi, gh, bh,
                                             gc, bc, ys, hT, cT);
  (void)in_sizes; (void)n_in; (void)out_size; (void)d_ws; (void)ws_size;
}
